// DynamicNeuralGraph_57758720197073
// MI455X (gfx1250) — matrix operand order repaired and checked by reading, not compiled, not run on hardware
//
#include <hip/hip_runtime.h>
#include <hip/hip_bf16.h>

// ---------------------------------------------------------------------------
// Problem sizes (fixed by the reference)
//   x: (B=256, I=784) f32      W: (N=256, I=784, H=128) f32   b: (N=256, H=128)
//   edge_index: (2, E=4096) i32, edge_weights: (E,) f32
//   out: (B=256, H=128) f32
// ---------------------------------------------------------------------------
#define NB 256   // neurons
#define II 784   // input dim
#define HH 128   // hidden dim
#define BB 256   // batch
#define EE 4096  // edges

typedef __attribute__((ext_vector_type(16))) __bf16 v16bf;
typedef __attribute__((ext_vector_type(8)))  float  v8f;
typedef __attribute__((ext_vector_type(4)))  float  v4f;

// ---------------------------------------------------------------------------
// Kernel 1: h0[n,b,h] = sum_i x[b,i] * W[n,i,h] + bias[n,h]
// One wave per 16x16 (B x H) tile of one neuron n, bf16 WMMA with f32 accum.
// Block = 128 threads = 4 waves covering 4 consecutive B-tiles.
// Grid = (B/16/4, H/16, N) = (4, 8, 256)
// ---------------------------------------------------------------------------
__global__ __launch_bounds__(128)
void h0_wmma_kernel(const float* __restrict__ x,
                    const float* __restrict__ W,
                    const float* __restrict__ bias,
                    float* __restrict__ h0)
{
    const int lane  = threadIdx.x & 31;
    const int wave  = threadIdx.x >> 5;
    const int half  = lane >> 4;        // 0: lanes 0-15, 1: lanes 16-31
    const int l16   = lane & 15;

    const int n     = blockIdx.z;
    const int btile = (blockIdx.x * 4 + wave) * 16;
    const int htile = blockIdx.y * 16;

    // A-fragment source: row of x for this lane (A 16x32 layout: M = lane%16)
    const float* __restrict__ Xrow = x + (size_t)(btile + l16) * II;
    // B-fragment source: column of W[n] for this lane (N = lane%16), stride H
    const float* __restrict__ Wn   = W + (size_t)n * II * HH + htile + l16;

    // C init with bias (bias depends only on the column -> same for all 8 rows)
    const float bv = bias[n * HH + htile + l16];
    v8f acc;
#pragma unroll
    for (int r = 0; r < 8; ++r) acc[r] = bv;

    const int koffA = half * 8;   // A: lanes 16-31 hold K base +8
    const int koffB = half * 8;   // B: lanes 16-31 hold K base +8

    // ---- main K loop: 24 full 16x16x32 steps (K = 0..767) ----
#pragma unroll 4
    for (int kb = 0; kb < 768; kb += 32) {
        // A: two runs of 8 contiguous f32 per lane -> four aligned b128 loads
        const v4f* X0 = (const v4f*)(Xrow + kb + koffA);        // K+0..7
        const v4f* X1 = (const v4f*)(Xrow + kb + koffA + 16);   // K+16..23
        v4f a0 = X0[0], a1 = X0[1], a2 = X1[0], a3 = X1[1];

        v16bf a;
#pragma unroll
        for (int j = 0; j < 4; ++j) {
            a[j]      = (__bf16)a0[j];
            a[j + 4]  = (__bf16)a1[j];
            a[j + 8]  = (__bf16)a2[j];
            a[j + 12] = (__bf16)a3[j];
        }

        v16bf bm;
#pragma unroll
        for (int j = 0; j < 16; ++j)
            bm[j] = (__bf16)Wn[(size_t)(kb + koffB + (j < 8 ? j : j + 8)) * HH];

        acc = __builtin_amdgcn_wmma_f32_16x16x32_bf16(
                  false, a, false, bm, (short)0, acc, false, false);
    }

    // ---- tail: K = 768..783 (16 valid values), fully branch-free ----
    // A: both lane halves read valid K (768+koffA+j <= 783); upper 8 -> zero.
    // B: same K order as A: elements 0-7 hold K=768+koffB+j (valid); upper 8
    //    (K>=784) -> zero.
    {
        v16bf a;
#pragma unroll
        for (int j = 0; j < 8; ++j) {
            a[j]     = (__bf16)Xrow[768 + koffA + j];
            a[j + 8] = (__bf16)0.0f;
        }
        v16bf bm;
#pragma unroll
        for (int j = 0; j < 16; ++j)
            bm[j] = (j < 8) ? (__bf16)Wn[(size_t)(768 + koffB + j) * HH] : (__bf16)0.0f;

        acc = __builtin_amdgcn_wmma_f32_16x16x32_bf16(
                  false, a, false, bm, (short)0, acc, false, false);
    }

    // ---- store D: VGPR r -> row M = r + half*8, col = l16 ----
    float* __restrict__ o =
        h0 + (size_t)n * (BB * HH) + (size_t)btile * HH + htile + l16;
#pragma unroll
    for (int r = 0; r < 8; ++r)
        o[(size_t)(r + half * 8) * HH] = acc[r];
}

// ---------------------------------------------------------------------------
// Kernel 2: sequential edge scan + mean over neurons.
// For a fixed (b,h) the scan only mixes the N=256 values of that element,
// so each thread owns one (b,h) and its private 256-float state in LDS.
// Stride 257 floats -> bank-conflict-free (257 mod 64 == 1).
// Edge (s,t,w) are wave-uniform -> scalar loads.
// Block = 32 threads (1 wave), 32.9 KB static LDS; grid = B*H/32 = 1024.
// ---------------------------------------------------------------------------
#define STRIDE 257
__global__ __launch_bounds__(32)
void edge_scan_kernel(const float* __restrict__ h0,
                      const int* __restrict__ eidx,
                      const float* __restrict__ ew,
                      float* __restrict__ out)
{
    __shared__ float st[32 * STRIDE];   // 32,896 bytes

    const int tid = threadIdx.x;
    const int gid = blockIdx.x * 32 + tid;   // 0 .. B*H-1
    const int b   = gid >> 7;                // / HH
    const int hh  = gid & (HH - 1);

    float* __restrict__ s = st + tid * STRIDE;

    // gather this element's state across all neurons (coalesced per n-slice)
    const float* __restrict__ src = h0 + (size_t)b * HH + hh;
#pragma unroll 4
    for (int n = 0; n < NB; ++n)
        s[n] = src[(size_t)n * (BB * HH)];

    // sequential, order-dependent scan over edges (faithful to lax.scan)
    const int* __restrict__ sIdx = eidx;        // edge_index[0, :]
    const int* __restrict__ tIdx = eidx + EE;   // edge_index[1, :]
    for (int e = 0; e < EE; ++e) {
        const int   ss = sIdx[e];
        const int   tt = tIdx[e];
        const float w  = ew[e];
        s[tt] += s[ss] * w;                     // ds_load -> fma -> ds_store
    }

    // mean over neurons
    float sum = 0.0f;
#pragma unroll 4
    for (int n = 0; n < NB; ++n) sum += s[n];
    out[gid] = sum * (1.0f / (float)NB);
}

// ---------------------------------------------------------------------------
extern "C" void kernel_launch(void* const* d_in, const int* in_sizes, int n_in,
                              void* d_out, int out_size, void* d_ws, size_t ws_size,
                              hipStream_t stream)
{
    const float* x    = (const float*)d_in[0];   // (B, I)
    const float* W    = (const float*)d_in[1];   // (N, I, H)
    const float* bias = (const float*)d_in[2];   // (N, H)
    const int*   eidx = (const int*)d_in[3];     // (2, E)
    const float* ew   = (const float*)d_in[4];   // (E,)
    float*       out  = (float*)d_out;           // (B, H)

    // scratch: h0 (N, B, H) f32 = 33.55 MB
    float* h0 = (float*)d_ws;

    dim3 gridA(BB / 16 / 4, HH / 16, NB);        // (4, 8, 256)
    h0_wmma_kernel<<<gridA, 128, 0, stream>>>(x, W, bias, h0);

    edge_scan_kernel<<<(BB * HH) / 32, 32, 0, stream>>>(h0, eidx, ew, out);
}
